// EncoderDecoder_20787641712983
// MI455X (gfx1250) — matrix operand order repaired and checked by reading, not compiled, not run on hardware
//
#include <hip/hip_runtime.h>
#include <hip/hip_bf16.h>
#include <stdint.h>
#include <stddef.h>

// Problem constants
#define B_  256
#define T_  512
#define U_  64
#define H_  512
#define SX_ 64
#define SY_ 32
#define G3H_ (3 * H_)

typedef __attribute__((ext_vector_type(16))) _Float16 v16h;
typedef __attribute__((ext_vector_type(8)))  _Float16 v8h;
typedef __attribute__((ext_vector_type(8)))  float    v8f;

// ---------------------------------------------------------------------------
// WMMA fragment loaders (gfx1250 wave32 layouts, CDNA5 ISA §7.12.2)
// A: 16x32 f16, row M = lane&15; lane<16 holds K {0..7,16..23}, lane>=16 holds
//    K {8..15,24..31} (2 halves per VGPR).
// B: 32x16 f16 (we feed W[N,K] row-major == B^T), col N = lane&15; lane<16
//    holds K {0..7,16..23}, lane>=16 holds K {8..15,24..31} (same as A).
// C/D: f32 16x16, VGPR v holds M = v + 8*(lane>=16), N = lane&15.
// ---------------------------------------------------------------------------
__device__ __forceinline__ v16h load_a16x32(const _Float16* base, int lda, int lane) {
  const int m  = lane & 15;
  const int kb = (lane & 16) ? 8 : 0;
  const _Float16* p = base + (size_t)m * (size_t)lda + kb;
  v8h lo = *(const v8h*)(p);        // K = kb .. kb+7
  v8h hi = *(const v8h*)(p + 16);   // K = kb+16 .. kb+23
  v16h r;
#pragma unroll
  for (int i = 0; i < 8; ++i) { r[i] = lo[i]; r[i + 8] = hi[i]; }
  return r;
}

__device__ __forceinline__ v16h load_b32x16(const _Float16* base, int ldw, int lane) {
  const int n  = lane & 15;
  const int kb = (lane & 16) ? 8 : 0;
  const _Float16* p = base + (size_t)n * (size_t)ldw + kb;
  v8h lo = *(const v8h*)(p);
  v8h hi = *(const v8h*)(p + 16);
  v16h r;
#pragma unroll
  for (int i = 0; i < 8; ++i) { r[i] = lo[i]; r[i + 8] = hi[i]; }
  return r;
}

__device__ __forceinline__ v8f wmma_f16(v16h a, v16h b, v8f c) {
  return __builtin_amdgcn_wmma_f32_16x16x32_f16(false, a, false, b, (short)0, c,
                                                false, false);
}

// ---------------------------------------------------------------------------
// One GRU step job description (kernel args passed by value).
// ---------------------------------------------------------------------------
struct GruJob {
  const _Float16* X; int ldx; int in_sz;
  const _Float16* Wih;              // [3H, in_sz] row-major f16
  const _Float16* Whh;              // [3H, H] row-major f16
  const float* bih; const float* bhh;
  const float* h_in32; const _Float16* h_in16;
  float* h_out32; _Float16* h_out16;
  _Float16* ys;                     // nullable; row stride 2H
  int ys_coloff;
};

// Fused GRU step body: wave `wv` (0..511) owns a 16(b) x 16(j) tile of h_new.
// gates = x_t @ Wih^T + h @ Whh^T; r,z accumulate combined; gate-n keeps the
// input and hidden parts separate (n = tanh(inn + r*hn)).
__device__ __forceinline__ void gru_step_body(const GruJob& J, int wv, int lane) {
  const int tm = wv & (B_ / 16 - 1);   // 0..15  (batch tile)
  const int tn = wv >> 4;              // 0..31  (hidden tile)
  const int c0 = tn * 16;

  v8f acc_r = {}, acc_z = {}, acc_in = {}, acc_hn = {};

  // Input-side GEMM (K = in_sz)
  const _Float16* Abase = J.X + (size_t)(tm * 16) * (size_t)J.ldx;
  const int in_sz = J.in_sz;
  for (int k = 0; k < in_sz; k += 32) {
    v16h a  = load_a16x32(Abase + k, J.ldx, lane);
    v16h br = load_b32x16(J.Wih + (size_t)(c0)          * in_sz + k, in_sz, lane);
    v16h bz = load_b32x16(J.Wih + (size_t)(c0 + H_)     * in_sz + k, in_sz, lane);
    v16h bn = load_b32x16(J.Wih + (size_t)(c0 + 2 * H_) * in_sz + k, in_sz, lane);
    acc_r  = wmma_f16(a, br, acc_r);
    acc_z  = wmma_f16(a, bz, acc_z);
    acc_in = wmma_f16(a, bn, acc_in);
  }
  // Hidden-side GEMM (K = H)
  const _Float16* Hbase = J.h_in16 + (size_t)(tm * 16) * H_;
  for (int k = 0; k < H_; k += 32) {
    v16h a  = load_a16x32(Hbase + k, H_, lane);
    v16h br = load_b32x16(J.Whh + (size_t)(c0)          * H_ + k, H_, lane);
    v16h bz = load_b32x16(J.Whh + (size_t)(c0 + H_)     * H_ + k, H_, lane);
    v16h bn = load_b32x16(J.Whh + (size_t)(c0 + 2 * H_) * H_ + k, H_, lane);
    acc_r  = wmma_f16(a, br, acc_r);
    acc_z  = wmma_f16(a, bz, acc_z);
    acc_hn = wmma_f16(a, bn, acc_hn);
  }

  const int col    = c0 + (lane & 15);
  const int rowoff = (lane & 16) ? 8 : 0;
  const float br_  = J.bih[col] + J.bhh[col];
  const float bz_  = J.bih[col + H_] + J.bhh[col + H_];
  const float bin_ = J.bih[col + 2 * H_];
  const float bhn_ = J.bhh[col + 2 * H_];
#pragma unroll
  for (int v = 0; v < 8; ++v) {
    const int b = tm * 16 + rowoff + v;
    const float r  = 1.0f / (1.0f + expf(-(acc_r[v] + br_)));
    const float z  = 1.0f / (1.0f + expf(-(acc_z[v] + bz_)));
    const float n  = tanhf(acc_in[v] + bin_ + r * (acc_hn[v] + bhn_));
    const float hp = J.h_in32[(size_t)b * H_ + col];
    const float hn = (1.0f - z) * n + z * hp;
    J.h_out32[(size_t)b * H_ + col] = hn;
    J.h_out16[(size_t)b * H_ + col] = (_Float16)hn;
    if (J.ys) J.ys[(size_t)b * (2 * H_) + J.ys_coloff + col] = (_Float16)hn;
  }
}

// Single-job step (decoder): grid = 64 blocks x 256 threads = 512 waves.
__global__ __launch_bounds__(256) void gru_step_kernel(GruJob J) {
  const int lane = threadIdx.x & 31;
  const int wv   = blockIdx.x * (blockDim.x >> 5) + (threadIdx.x >> 5);
  gru_step_body(J, wv, lane);
}

// Dual-job step (bidirectional encoder: fwd + bwd in one launch).
// Grid = 128 blocks x 256 threads = 1024 waves; blocks [0,64) run J0,
// blocks [64,128) run J1.  Doubles occupancy and halves serialized launches.
__global__ __launch_bounds__(256) void gru_step_dual_kernel(GruJob J0, GruJob J1) {
  const int lane = threadIdx.x & 31;
  const int blk  = blockIdx.x & 63;
  const int wv   = blk * (blockDim.x >> 5) + (threadIdx.x >> 5);
  if (blockIdx.x < 64) gru_step_body(J0, wv, lane);
  else                 gru_step_body(J1, wv, lane);
}

// ---------------------------------------------------------------------------
// C[M,N](f32,+f16) = A[M,K](f16) @ W[N,K]^T(f16) + bias.  One wave per tile.
// Grid must supply (M/16)*(N/16) waves.
// ---------------------------------------------------------------------------
__global__ __launch_bounds__(256) void gemm_bias_kernel(
    const _Float16* __restrict__ A, int lda,
    const _Float16* __restrict__ W, int ldw, const float* __restrict__ bias,
    float* __restrict__ C32, _Float16* __restrict__ C16, int N, int K) {
  const int lane = threadIdx.x & 31;
  const int wave = blockIdx.x * (blockDim.x >> 5) + (threadIdx.x >> 5);
  const int ntiles = N >> 4;
  const int tm = wave / ntiles;
  const int tn = wave % ntiles;
  v8f acc = {};
  const _Float16* Abase = A + (size_t)(tm * 16) * (size_t)lda;
  const _Float16* Wbase = W + (size_t)(tn * 16) * (size_t)ldw;
  for (int k = 0; k < K; k += 32) {
    v16h a = load_a16x32(Abase + k, lda, lane);
    v16h b = load_b32x16(Wbase + k, ldw, lane);
    acc = wmma_f16(a, b, acc);
  }
  const int col    = tn * 16 + (lane & 15);
  const int rowoff = (lane & 16) ? 8 : 0;
  const float bb   = bias[col];
#pragma unroll
  for (int v = 0; v < 8; ++v) {
    const int m = tm * 16 + rowoff + v;
    const float x = acc[v] + bb;
    C32[(size_t)m * N + col] = x;
    if (C16) C16[(size_t)m * N + col] = (_Float16)x;
  }
}

// ---------------------------------------------------------------------------
// Decoder logits + log_softmax + dot(target):  out[b] += sum_c logp[b,c]*y[b,u,c]
// One wave handles 16 rows x 32 logit columns (two WMMA accumulators).
// Row-wise reductions via half-wave shfl_xor (masks 1,2,4,8 stay in group).
// Launch: 1 block of 512 threads (16 waves).
// ---------------------------------------------------------------------------
__global__ __launch_bounds__(512) void dec_logits_kernel(
    const _Float16* __restrict__ h16,          // [B,H] (post-step hidden)
    const _Float16* __restrict__ Wlin,         // [SY,H] row-major f16
    const float* __restrict__ blin,            // [SY]
    const float* __restrict__ y,               // [B,U,SY] f32 targets
    int u, float* __restrict__ out) {
  const int lane = threadIdx.x & 31;
  const int tm   = threadIdx.x >> 5;           // 0..15
  v8f acc0 = {}, acc1 = {};
  const _Float16* Abase = h16 + (size_t)(tm * 16) * H_;
  for (int k = 0; k < H_; k += 32) {
    v16h a  = load_a16x32(Abase + k, H_, lane);
    v16h b0 = load_b32x16(Wlin + k, H_, lane);                    // cols 0..15
    v16h b1 = load_b32x16(Wlin + (size_t)16 * H_ + k, H_, lane);  // cols 16..31
    acc0 = wmma_f16(a, b0, acc0);
    acc1 = wmma_f16(a, b1, acc1);
  }
  const int c      = lane & 15;
  const int rowoff = (lane & 16) ? 8 : 0;
  const float bb0 = blin[c], bb1 = blin[c + 16];
  float contrib[8];
#pragma unroll
  for (int v = 0; v < 8; ++v) {
    float l0 = acc0[v] + bb0;
    float l1 = acc1[v] + bb1;
    float m = fmaxf(l0, l1);
#pragma unroll
    for (int mask = 1; mask <= 8; mask <<= 1) m = fmaxf(m, __shfl_xor(m, mask, 32));
    float s = expf(l0 - m) + expf(l1 - m);
#pragma unroll
    for (int mask = 1; mask <= 8; mask <<= 1) s += __shfl_xor(s, mask, 32);
    const float lse = m + logf(s);
    const int b = tm * 16 + rowoff + v;
    const float yc0 = y[((size_t)b * U_ + u) * SY_ + c];
    const float yc1 = y[((size_t)b * U_ + u) * SY_ + c + 16];
    float t = (l0 - lse) * yc0 + (l1 - lse) * yc1;
#pragma unroll
    for (int mask = 1; mask <= 8; mask <<= 1) t += __shfl_xor(t, mask, 32);
    contrib[v] = t;
  }
  if ((lane & 15) == 0) {   // unique writer per (b) per launch -> deterministic
#pragma unroll
    for (int v = 0; v < 8; ++v) out[tm * 16 + rowoff + v] += contrib[v];
  }
}

// ------------------------------- utilities ---------------------------------
__global__ void cvt_f16_kernel(const float* __restrict__ s, _Float16* __restrict__ d, int n) {
  for (int i = blockIdx.x * blockDim.x + threadIdx.x; i < n; i += gridDim.x * blockDim.x)
    d[i] = (_Float16)s[i];
}
__global__ void zero_f32_kernel(float* p, int n) {
  for (int i = blockIdx.x * blockDim.x + threadIdx.x; i < n; i += gridDim.x * blockDim.x)
    p[i] = 0.0f;
}
__global__ void zero_f16_kernel(_Float16* p, int n) {
  for (int i = blockIdx.x * blockDim.x + threadIdx.x; i < n; i += gridDim.x * blockDim.x)
    p[i] = (_Float16)0.0f;
}
__global__ void pack_enc_kernel(const float* __restrict__ h32, _Float16* __restrict__ enc,
                                int coloff) {
  int i = blockIdx.x * blockDim.x + threadIdx.x;
  if (i < B_ * H_) {
    int b = i / H_, c = i % H_;
    enc[(size_t)b * (2 * H_) + coloff + c] = (_Float16)h32[i];
  }
}

// ---------------------------------------------------------------------------
extern "C" void kernel_launch(void* const* d_in, const int* in_sizes, int n_in,
                              void* d_out, int out_size, void* d_ws, size_t ws_size,
                              hipStream_t stream) {
  (void)in_sizes; (void)n_in; (void)out_size; (void)ws_size;
  const float* x = (const float*)d_in[0];
  const float* y = (const float*)d_in[1];
  // GRU params: [0]=enc0f [1]=enc0b [2]=enc1f [3]=enc1b [4]=decoder
  const float *W_ih[5], *W_hh[5], *b_ih[5], *b_hh[5];
  int idx = 2;
  for (int i = 0; i < 4; ++i) {
    W_ih[i] = (const float*)d_in[idx++]; W_hh[i] = (const float*)d_in[idx++];
    b_ih[i] = (const float*)d_in[idx++]; b_hh[i] = (const float*)d_in[idx++];
  }
  const float* enc_lin_W = (const float*)d_in[18];
  const float* enc_lin_b = (const float*)d_in[19];
  W_ih[4] = (const float*)d_in[20]; W_hh[4] = (const float*)d_in[21];
  b_ih[4] = (const float*)d_in[22]; b_hh[4] = (const float*)d_in[23];
  const float* dec_lin_W = (const float*)d_in[24];
  const float* dec_lin_b = (const float*)d_in[25];
  float* out = (float*)d_out;

  // ---- workspace carve (~303 MB) ----
  uintptr_t cur = (uintptr_t)d_ws;
  auto carve = [&](size_t bytes) -> void* {
    void* p = (void*)cur; cur += (bytes + 255) & ~(size_t)255; return p;
  };
  const int insz[5] = {SX_, SX_, 2 * H_, 2 * H_, SY_};
  _Float16* xh = (_Float16*)carve((size_t)B_ * T_ * SX_ * 2);
  _Float16* yh = (_Float16*)carve((size_t)B_ * U_ * SY_ * 2);
  _Float16 *wih16[5], *whh16[5];
  for (int i = 0; i < 5; ++i) {
    wih16[i] = (_Float16*)carve((size_t)G3H_ * insz[i] * 2);
    whh16[i] = (_Float16*)carve((size_t)G3H_ * H_ * 2);
  }
  _Float16* wlin16  = (_Float16*)carve((size_t)H_ * 2 * H_ * 2);
  _Float16* wdlin16 = (_Float16*)carve((size_t)SY_ * H_ * 2);
  _Float16* ys0     = (_Float16*)carve((size_t)T_ * B_ * 2 * H_ * 2);  // layer-0 output seq
  _Float16* ench    = (_Float16*)carve((size_t)B_ * 2 * H_ * 2);
  // Two independent ping-pong hidden-state sets (fwd stream F, bwd stream G).
  _Float16* h16[4]; float* h32[4];
  for (int i = 0; i < 4; ++i) h16[i] = (_Float16*)carve((size_t)B_ * H_ * 2);
  _Float16* zrow = (_Float16*)carve(64 * 2);
  for (int i = 0; i < 4; ++i) h32[i] = (float*)carve((size_t)B_ * H_ * 4);

  auto cvt = [&](const float* s, _Float16* d, size_t n) {
    int blocks = (int)((n + 255) / 256); if (blocks > 2048) blocks = 2048;
    cvt_f16_kernel<<<blocks, 256, 0, stream>>>(s, d, (int)n);
  };
  cvt(x, xh, (size_t)B_ * T_ * SX_);
  cvt(y, yh, (size_t)B_ * U_ * SY_);
  for (int i = 0; i < 5; ++i) {
    cvt(W_ih[i], wih16[i], (size_t)G3H_ * insz[i]);
    cvt(W_hh[i], whh16[i], (size_t)G3H_ * H_);
  }
  cvt(enc_lin_W, wlin16, (size_t)H_ * 2 * H_);
  cvt(dec_lin_W, wdlin16, (size_t)SY_ * H_);
  zero_f16_kernel<<<1, 64, 0, stream>>>(zrow, 64);

  // hidden-state ping-pong indices: F uses {0,1}, G uses {2,3}
  int fc = 0, fn = 1, gc = 2, gn = 3;
  auto zero_all_h = [&]() {
    for (int i = 0; i < 4; ++i) {
      zero_f32_kernel<<<512, 256, 0, stream>>>(h32[i], B_ * H_);
      zero_f16_kernel<<<512, 256, 0, stream>>>(h16[i], B_ * H_);
    }
  };
  auto mkjob = [&](const _Float16* X, int ldx, int in_sz, int wi,
                   int hin, int hout, _Float16* ys, int coloff) {
    GruJob J;
    J.X = X; J.ldx = ldx; J.in_sz = in_sz;
    J.Wih = wih16[wi]; J.Whh = whh16[wi]; J.bih = b_ih[wi]; J.bhh = b_hh[wi];
    J.h_in32 = h32[hin]; J.h_in16 = h16[hin];
    J.h_out32 = h32[hout]; J.h_out16 = h16[hout];
    J.ys = ys; J.ys_coloff = coloff;
    return J;
  };
  auto swap2 = [](int& a, int& b) { int t = a; a = b; b = t; };

  // ---- encoder layer 0: fwd (ys cols [0,H)) + bwd (ys cols [H,2H)) fused ----
  zero_all_h();
  for (int t = 0; t < T_; ++t) {
    const int tb = T_ - 1 - t;
    GruJob JF = mkjob(xh + (size_t)t  * SX_, T_ * SX_, SX_, 0, fc, fn,
                      ys0 + (size_t)t  * B_ * 2 * H_, 0);
    GruJob JB = mkjob(xh + (size_t)tb * SX_, T_ * SX_, SX_, 1, gc, gn,
                      ys0 + (size_t)tb * B_ * 2 * H_, H_);
    gru_step_dual_kernel<<<128, 256, 0, stream>>>(JF, JB);
    swap2(fc, fn); swap2(gc, gn);
  }
  // ---- encoder layer 1: fwd + bwd fused (input = full ys0) ----
  zero_all_h();
  for (int t = 0; t < T_; ++t) {
    const int tb = T_ - 1 - t;
    GruJob JF = mkjob(ys0 + (size_t)t  * B_ * 2 * H_, 2 * H_, 2 * H_, 2, fc, fn,
                      nullptr, 0);
    GruJob JB = mkjob(ys0 + (size_t)tb * B_ * 2 * H_, 2 * H_, 2 * H_, 3, gc, gn,
                      nullptr, 0);
    gru_step_dual_kernel<<<128, 256, 0, stream>>>(JF, JB);
    swap2(fc, fn); swap2(gc, gn);
  }
  // enc = [hT_b, hT_f]
  pack_enc_kernel<<<512, 256, 0, stream>>>(h32[fc], ench, H_);
  pack_enc_kernel<<<512, 256, 0, stream>>>(h32[gc], ench, 0);

  // ---- dec_h0 = enc @ enc_lin_W^T + b  (M=256,N=512,K=1024) -> F-current ----
  gemm_bias_kernel<<<64, 256, 0, stream>>>(ench, 2 * H_, wlin16, 2 * H_, enc_lin_b,
                                           h32[fc], h16[fc], H_, 2 * H_);

  // ---- decoder: 64 steps of GRU + fused logits/log-softmax/dot ----
  zero_f32_kernel<<<1, 256, 0, stream>>>(out, B_);
  for (int u = 0; u < U_; ++u) {
    const _Float16* Xr; int ldx;
    if (u == 0) { Xr = zrow; ldx = 0; }                       // y_prev[0] = zeros
    else        { Xr = yh + (size_t)(u - 1) * SY_; ldx = U_ * SY_; }
    GruJob JD = mkjob(Xr, ldx, SY_, 4, fc, fn, nullptr, 0);
    gru_step_kernel<<<64, 256, 0, stream>>>(JD);
    swap2(fc, fn);
    dec_logits_kernel<<<1, 512, 0, stream>>>(h16[fc], wdlin16, dec_lin_b, y, u, out);
  }
}
